// FuseAttention_37993280700897
// MI455X (gfx1250) — compile-verified, matrix operand order checked by reading, not run on hardware
//
#include <hip/hip_runtime.h>

// ---------------------------------------------------------------------------
// CDNA5 / gfx1250 fused PVT spatial-reduction attention.
// bf16 WMMA for all matmuls + Tensor Data Mover (TDM) for LDS tile staging.
// ---------------------------------------------------------------------------

typedef __attribute__((ext_vector_type(16))) __bf16 v16bf;
typedef __attribute__((ext_vector_type(8)))  float  v8f;
typedef __attribute__((ext_vector_type(4)))  unsigned int v4u;
typedef __attribute__((ext_vector_type(8)))  int    v8i;
typedef __attribute__((ext_vector_type(4)))  int    v4i;
typedef unsigned short u16;
typedef unsigned int   u32;
typedef unsigned long long u64;

__device__ __forceinline__ u16 f2bf(float f) {
    union { float f; u32 u; } c; c.f = f;
    u32 u = c.u;
    u32 r = (u + 0x7FFFu + ((u >> 16) & 1u)) >> 16;   // round-to-nearest-even
    return (u16)r;
}

__device__ __forceinline__ v16bf make_frag(uint4 lo, uint4 hi) {
    union { uint4 u[2]; v16bf v; } c;
    c.u[0] = lo; c.u[1] = hi;
    return c.v;
}

// ---------------------------------------------------------------------------
// Tensor Data Mover: 2D (or 1D when tile_rows==0) bf16 tile, global -> LDS.
//   lds_off        : byte offset of destination inside LDS
//   gaddr          : global address of tile start
//   remaining_rows : tensor_dim1 (rows past this return zero -> free bounds)
//   row_stride     : tensor_dim0_stride in elements
//   tile_rows      : tile_dim1 (0 => 1-D tile)
//   tile_cols      : tile_dim0 in elements
// D# packing per CDNA5 ISA 8.3/8.4: group0 = {flags, lds_addr, global_addr,
// type=2}; group1 = {data_size=2B, tensor dims, tile dims, strides}.
// This toolchain exposes the 6-arg builtin (g0, g1, g2, g3, g4, cpol).
// ---------------------------------------------------------------------------
__device__ __forceinline__ void tdm_load_2d(
    u32 lds_off, const u16* gaddr, u32 remaining_rows, u32 row_stride,
    u32 tile_rows, u32 tile_cols)
{
    u64 ga = (u64)(size_t)gaddr;
    v4u g0;
    g0[0] = 1u;                                          // count=1, user mode
    g0[1] = lds_off;                                     // lds_addr
    g0[2] = (u32)(ga & 0xFFFFFFFFu);                     // global_addr[31:0]
    g0[3] = (u32)((ga >> 32) & 0x01FFFFFFu) | (2u << 30); // addr[56:32] | type=2

    u32 w[8] = {0, 0, 0, 0, 0, 0, 0, 0};
    w[0] = (1u << 16);                                   // data_size = 2 bytes
    const u32 td0 = 0x7FFFFFFFu;                         // dim0 effectively unbounded
    w[1] |= (td0 << 16);                                 // tensor_dim0 @ bit 48
    w[2] |= (td0 >> 16);
    w[2] |= (remaining_rows << 16);                      // tensor_dim1 @ bit 80
    w[3] |= (remaining_rows >> 16);
    w[3] |= (tile_cols << 16);                           // tile_dim0 @ bit 112
    w[4]  = tile_rows & 0xFFFFu;                         // tile_dim1 (tile_dim2=0)
    w[5]  = row_stride;                                  // tensor_dim0_stride[31:0]
    // stride[47:32]=0, tensor_dim1_stride=0 (2-D)

    v8i g1;
#pragma unroll
    for (int i = 0; i < 8; i++) g1[i] = (int)w[i];
    v4i z4 = {0, 0, 0, 0};
    v8i z8 = {0, 0, 0, 0, 0, 0, 0, 0};
    __builtin_amdgcn_tensor_load_to_lds(g0, g1, z4, z4, z8, 0);
}

// ---------------------------------------------------------------------------
// Elementwise converts / init
// ---------------------------------------------------------------------------
__global__ void cvt_f32_bf16(const float* __restrict__ in, u16* __restrict__ out, int n) {
    int i = blockIdx.x * blockDim.x + threadIdx.x;
    int stride = gridDim.x * blockDim.x;
    for (; i < n; i += stride) out[i] = f2bf(in[i]);
}

// sr_w [O=256][I=256][KH][KW] f32  ->  [O][khw][I] bf16  (GEMM K index = khw*256 + i)
__global__ void cvt_srw(const float* __restrict__ in, u16* __restrict__ out, int KHW) {
    int n = 256 * 256 * KHW;
    int e = blockIdx.x * blockDim.x + threadIdx.x;
    int stride = gridDim.x * blockDim.x;
    for (; e < n; e += stride) {
        int i   = e & 255;
        int khw = (e >> 8) % KHW;
        int o   = e / (256 * KHW);
        out[e] = f2bf(in[(o * 256 + i) * KHW + khw]);
    }
}

__global__ void zero_u16(u16* __restrict__ p, int n) {
    int i = blockIdx.x * blockDim.x + threadIdx.x;
    int stride = gridDim.x * blockDim.x;
    for (; i < n; i += stride) p[i] = 0;
}

// im2col for the stride==kernel conv: x [B][H*W][256] bf16 -> A [B*49][KHW*256]
__global__ void gather_patches(const u16* __restrict__ x, u16* __restrict__ out,
                               int Wimg, int s, int KW, int KHW) {
    long n = (long)8 * 49 * KHW * 256;
    long e = (long)blockIdx.x * blockDim.x + threadIdx.x;
    long stride = (long)gridDim.x * blockDim.x;
    for (; e < n; e += stride) {
        int  i   = (int)(e & 255);
        int  khw = (int)((e >> 8) % KHW);
        long m   = e / (256 * (long)KHW);
        int  b   = (int)(m / 49);
        int  os  = (int)(m % 49);
        int  oy = os / 7, ox = os % 7;
        int  kh = khw / KW, kw = khw % KW;
        int  row = (oy * s + kh) * Wimg + ox * s + kw;
        out[e] = x[((long)b * (Wimg * Wimg) + row) * 256 + i];
    }
}

// ---------------------------------------------------------------------------
// Generic bf16 WMMA GEMM:  C[M,N] = A[M,K] * W[N,K]^T (+bias)
// Block = 256 threads = 8 waves, tile 128x128, K step 32.
// Tile staging via TDM (wave 0 issues descriptors, TENSORcnt-synced).
// Wave grid 2(m) x 4(n); each wave: 4x2 tiles of 16x16 -> 8 v8f accumulators.
// ---------------------------------------------------------------------------
#define GMT 128
#define GNT 128
#define GKT 32

__global__ __launch_bounds__(256) void gemm_bf16(
    const u16* __restrict__ A, const u16* __restrict__ W,
    const float* __restrict__ bias, void* __restrict__ outp,
    int M, int N, int K, int out_bf16)
{
    __shared__ u16 Asm[GMT * GKT];
    __shared__ u16 Wsm[GNT * GKT];

    const int tid  = threadIdx.x;
    const int lane = tid & 31;
    const int wave = tid >> 5;
    const int wm   = wave >> 2;          // 0..1
    const int wn   = wave & 3;           // 0..3
    const int m0   = blockIdx.y * GMT;
    const int n0   = blockIdx.x * GNT;
    const int hi   = lane >> 4;
    const int l16  = lane & 15;
    const int ks   = hi ? 8 : 0;         // A/B fragment K-offset per ISA layout

    const u32 asm_off = (u32)(size_t)(void*)Asm;   // low 32 bits = LDS offset
    const u32 wsm_off = (u32)(size_t)(void*)Wsm;
    const u32 remA = (u32)(M - m0);                // hardware zero-fill past M
    const u32 remW = (u32)(N - n0);

    v8f acc[4][2];
#pragma unroll
    for (int mt = 0; mt < 4; mt++)
#pragma unroll
        for (int nt = 0; nt < 2; nt++) { v8f z = {}; acc[mt][nt] = z; }

    for (int k0 = 0; k0 < K; k0 += GKT) {
        __syncthreads();                               // previous tile consumed
        if (wave == 0) {
            tdm_load_2d(asm_off, A + (long)m0 * K + k0, remA, (u32)K, GMT, GKT);
            tdm_load_2d(wsm_off, W + (long)n0 * K + k0, remW, (u32)K, GNT, GKT);
            __builtin_amdgcn_s_wait_tensorcnt((short)0);
        }
        __syncthreads();                               // tiles visible to all waves

        v16bf bfr[2];
#pragma unroll
        for (int nt = 0; nt < 2; nt++) {
            const u16* p = Wsm + (wn * 32 + nt * 16 + l16) * GKT;
            bfr[nt] = make_frag(*(const uint4*)(p + ks), *(const uint4*)(p + ks + 16));
        }
#pragma unroll
        for (int mt = 0; mt < 4; mt++) {
            const u16* p = Asm + (wm * 64 + mt * 16 + l16) * GKT;
            v16bf afr = make_frag(*(const uint4*)(p + ks), *(const uint4*)(p + ks + 16));
#pragma unroll
            for (int nt = 0; nt < 2; nt++)
                acc[mt][nt] = __builtin_amdgcn_wmma_f32_16x16x32_bf16(
                    false, afr, false, bfr[nt], (short)0, acc[mt][nt], false, false);
        }
    }

    // C layout: VGPR j -> row j + 8*hi, column = lane%16.
#pragma unroll
    for (int mt = 0; mt < 4; mt++) {
#pragma unroll
        for (int nt = 0; nt < 2; nt++) {
            int colg = n0 + wn * 32 + nt * 16 + l16;
            float bv = bias ? bias[colg] : 0.f;
#pragma unroll
            for (int j = 0; j < 8; j++) {
                int rowg = m0 + wm * 64 + mt * 16 + j + 8 * hi;
                if (rowg < M) {
                    float v = acc[mt][nt][j] + bv;
                    if (out_bf16) ((u16*)outp)[(long)rowg * N + colg] = f2bf(v);
                    else          ((float*)outp)[(long)rowg * N + colg] = v;
                }
            }
        }
    }
}

// ---------------------------------------------------------------------------
// LayerNorm + KV projection (small: M=392, N=512, K=256; fp32 exact).
// Writes K/V in padded bf16 layout [B][NH][112][32].
// ---------------------------------------------------------------------------
__global__ __launch_bounds__(256) void ln_kv(
    const float* __restrict__ xr, const float* __restrict__ srb,
    const float* __restrict__ nw, const float* __restrict__ nb,
    const float* __restrict__ kvw,
    u16* __restrict__ Kmat, u16* __restrict__ Vmat, int tokoff)
{
    __shared__ float xsh[256];
    const int row = blockIdx.x;          // 0 .. B*49-1
    const int b = row / 49, os = row % 49;
    const int t = threadIdx.x;

    float x = xr[row * 256 + t] + srb[t];
    xsh[t] = x;
    __syncthreads();
    float mu = 0.f;
    for (int c = 0; c < 256; c++) mu += xsh[c];
    mu *= (1.f / 256.f);
    float var = 0.f;
    for (int c = 0; c < 256; c++) { float d = xsh[c] - mu; var += d * d; }
    var *= (1.f / 256.f);
    float inv = rsqrtf(var + 1e-5f);
    __syncthreads();
    xsh[t] = (x - mu) * inv * nw[t] + nb[t];
    __syncthreads();

    const int tok = tokoff + os;
#pragma unroll
    for (int half = 0; half < 2; half++) {
        int j = t + half * 256;                  // 0..511
        const float* wrow = kvw + (long)j * 256;
        float s = 0.f;
        for (int c = 0; c < 256; c++) s += xsh[c] * wrow[c];
        int jj = j & 255;
        int h = jj >> 5, d = jj & 31;
        u16* dst = (j < 256) ? Kmat : Vmat;
        dst[(((long)b * 8 + h) * 112 + tok) * 32 + d] = f2bf(s);
    }
}

// ---------------------------------------------------------------------------
// Fused attention: one wave per (b, h, 16-query tile). Nk=98 padded to 112.
// K tile staged via TDM; QK^T (7 WMMAs) -> exact softmax (shfl_xor row
// reductions over 16-lane groups) -> PV (8 WMMAs).
// ---------------------------------------------------------------------------
#define NKP 112

__global__ __launch_bounds__(32) void attention(
    const u16* __restrict__ Qb0, const u16* __restrict__ Qb1,
    const u16* __restrict__ Kmat, const u16* __restrict__ Vmat,
    u16* __restrict__ Xb)
{
    __shared__ u16 Ksm[NKP * 32];      // [tok][d]
    __shared__ u16 Vtsm[32 * 128];     // [d][tok], tok padded to 128
    __shared__ u16 Psm[16 * 128];      // softmax probs, row-major

    const int lane = threadIdx.x;
    const int qt = blockIdx.x;         // 0..244 (16 queries each)
    const int h  = blockIdx.y;
    const int b  = blockIdx.z;
    const int hi = lane >> 4, l16 = lane & 15;
    const int ks = hi ? 8 : 0;

    const u16* Kg = Kmat + (((long)b * 8 + h) * NKP) * 32;
    const u16* Vg = Vmat + (((long)b * 8 + h) * NKP) * 32;

    // K: one 1-D TDM descriptor (7168 contiguous bytes)
    tdm_load_2d((u32)(size_t)(void*)Ksm, Kg, 1u, NKP * 32, 0, NKP * 32);

    for (int i = lane; i < 32 * 128; i += 32) {          // transpose V into [d][tok]
        int d = i >> 7, tok = i & 127;
        Vtsm[d * 128 + tok] = (tok < NKP) ? Vg[tok * 32 + d] : (u16)0;
    }
    if (lane < 16)                                       // zero P pad cols 112..127
        for (int r = 0; r < 16; r++) Psm[r * 128 + 112 + lane] = 0;

    __builtin_amdgcn_s_wait_tensorcnt((short)0);
    __syncthreads();

    // Q fragment (A matrix, 16x32): each 16-query tile lies entirely in one branch.
    const int qbase = qt * 16;
    const u16* qrow;
    {
        int q = qbase + l16;
        if (q < 3136) qrow = Qb0 + (((long)b * 3136 + q) * 256) + h * 32;
        else          qrow = Qb1 + (((long)b * 784 + (q - 3136)) * 256) + h * 32;
    }
    v16bf qf = make_frag(*(const uint4*)(qrow + ks), *(const uint4*)(qrow + ks + 16));

    // S = Q K^T : 7 key tiles of 16
    v8f S[7];
#pragma unroll
    for (int t = 0; t < 7; t++) {
        const u16* p = Ksm + (t * 16 + l16) * 32;
        v16bf kf = make_frag(*(const uint4*)(p + ks), *(const uint4*)(p + ks + 16));
        v8f z = {};
        S[t] = __builtin_amdgcn_wmma_f32_16x16x32_bf16(
            false, qf, false, kf, (short)0, z, false, false);
    }

    // scale + mask + row max (rows: j + 8*hi; cols across 16-lane group x 7 tiles)
    const float scale = 0.17677669529663687f;   // 32^-0.5
    float rmax[8], rsum[8];
#pragma unroll
    for (int j = 0; j < 8; j++) { rmax[j] = -3.0e38f; rsum[j] = 0.f; }
#pragma unroll
    for (int t = 0; t < 7; t++) {
        bool valid = (t * 16 + l16) < 98;
#pragma unroll
        for (int j = 0; j < 8; j++) {
            float v = valid ? S[t][j] * scale : -3.0e38f;
            S[t][j] = v;
            rmax[j] = fmaxf(rmax[j], v);
        }
    }
#pragma unroll
    for (int j = 0; j < 8; j++)
        for (int off = 1; off < 16; off <<= 1)
            rmax[j] = fmaxf(rmax[j], __shfl_xor(rmax[j], off, 32));

    const int row0 = hi * 8;
#pragma unroll
    for (int t = 0; t < 7; t++)
#pragma unroll
        for (int j = 0; j < 8; j++) {
            float p = __expf(S[t][j] - rmax[j]);
            rsum[j] += p;
            Psm[(j + row0) * 128 + t * 16 + l16] = f2bf(p);
        }
#pragma unroll
    for (int j = 0; j < 8; j++)
        for (int off = 1; off < 16; off <<= 1)
            rsum[j] += __shfl_xor(rsum[j], off, 32);
    __syncthreads();

    // O = P V : K = 128 (4 chunks of 32 keys), N = 32 (2 tiles of 16)
    v8f O[2]; { v8f z = {}; O[0] = z; O[1] = z; }
#pragma unroll
    for (int kc = 0; kc < 4; kc++) {
        const u16* pp = Psm + l16 * 128 + kc * 32;
        v16bf pf = make_frag(*(const uint4*)(pp + ks), *(const uint4*)(pp + ks + 16));
#pragma unroll
        for (int nt = 0; nt < 2; nt++) {
            const u16* vp = Vtsm + (nt * 16 + l16) * 128 + kc * 32;
            v16bf vf = make_frag(*(const uint4*)(vp + ks), *(const uint4*)(vp + ks + 16));
            O[nt] = __builtin_amdgcn_wmma_f32_16x16x32_bf16(
                false, pf, false, vf, (short)0, O[nt], false, false);
        }
    }

    // normalize and store X bf16 at [b][q][h*32+d] (concat layout -> final GEMM A)
#pragma unroll
    for (int nt = 0; nt < 2; nt++)
#pragma unroll
        for (int j = 0; j < 8; j++) {
            int q = qbase + j + 8 * hi;
            float v = O[nt][j] / rsum[j];
            Xb[(((long)b * 3920 + q) * 256) + h * 32 + nt * 16 + l16] = f2bf(v);
        }
}

// ---------------------------------------------------------------------------
// Host launcher
// ---------------------------------------------------------------------------
extern "C" void kernel_launch(void* const* d_in, const int* in_sizes, int n_in,
                              void* d_out, int out_size, void* d_ws, size_t ws_size,
                              hipStream_t stream) {
    (void)in_sizes; (void)n_in; (void)out_size; (void)ws_size;

    const float* x0      = (const float*)d_in[0];
    const float* x1      = (const float*)d_in[1];
    const float* q_w     = (const float*)d_in[2];
    const float* kv_w    = (const float*)d_in[3];
    const float* proj_w  = (const float*)d_in[4];
    const float* proj_b  = (const float*)d_in[5];
    const float* sr0_w   = (const float*)d_in[6];
    const float* sr0_b   = (const float*)d_in[7];
    const float* sr1_w   = (const float*)d_in[8];
    const float* sr1_b   = (const float*)d_in[9];
    const float* norm0_w = (const float*)d_in[10];
    const float* norm0_b = (const float*)d_in[11];
    const float* norm1_w = (const float*)d_in[12];
    const float* norm1_b = (const float*)d_in[13];

    char* wsp = (char*)d_ws;
    auto alloc = [&](size_t bytes) -> char* {
        char* p = wsp;
        wsp += (bytes + 255) & ~(size_t)255;
        return p;
    };

    u16*   x0b   = (u16*)alloc((size_t)6422528 * 2);   // 8*3136*256 bf16
    u16*   x1b   = (u16*)alloc((size_t)1605632 * 2);   // 8*784*256
    u16*   qwb   = (u16*)alloc((size_t)65536 * 2);
    u16*   pwb   = (u16*)alloc((size_t)65536 * 2);
    u16*   sr0wt = (u16*)alloc((size_t)4194304 * 2);   // 256*16384
    u16*   sr1wt = (u16*)alloc((size_t)1048576 * 2);   // 256*4096
    u16*   Qb0   = (u16*)alloc((size_t)6422528 * 2);
    u16*   Qb1   = (u16*)alloc((size_t)1605632 * 2);
    float* xr0   = (float*)alloc((size_t)100352 * 4);  // 392*256
    float* xr1   = (float*)alloc((size_t)100352 * 4);
    u16*   Kmat  = (u16*)alloc((size_t)229376 * 2);    // 8*8*112*32
    u16*   Vmat  = (u16*)alloc((size_t)229376 * 2);
    char*  agbase = wsp;                               // im2col region, reused as Xb
    u16*   Ag0   = (u16*)alloc((size_t)6422528 * 2);   // 392*16384
    u16*   Ag1   = (u16*)alloc((size_t)1605632 * 2);   // 392*4096
    u16*   Xb    = (u16*)agbase;                       // 8*3920*256 bf16 (aliases Ag0+Ag1)

    const int thr = 256;

    // 1. convert activations + plain weights to bf16
    cvt_f32_bf16<<<4096, thr, 0, stream>>>(x0, x0b, 6422528);
    cvt_f32_bf16<<<1024, thr, 0, stream>>>(x1, x1b, 1605632);
    cvt_f32_bf16<<<64,   thr, 0, stream>>>(q_w, qwb, 65536);
    cvt_f32_bf16<<<64,   thr, 0, stream>>>(proj_w, pwb, 65536);
    cvt_srw<<<4096, thr, 0, stream>>>(sr0_w, sr0wt, 64);
    cvt_srw<<<1024, thr, 0, stream>>>(sr1_w, sr1wt, 16);

    // 2. Q projection per branch (bf16 out, head-interleaved row layout)
    gemm_bf16<<<dim3(2, 25088 / 128), 256, 0, stream>>>(x0b, qwb, nullptr, Qb0, 25088, 256, 256, 1);
    gemm_bf16<<<dim3(2,  6272 / 128), 256, 0, stream>>>(x1b, qwb, nullptr, Qb1,  6272, 256, 256, 1);

    // 3. SR conv = im2col gather + WMMA GEMM (f32 out, bias added in LN)
    gather_patches<<<4096, thr, 0, stream>>>(x0b, Ag0, 56, 8, 8, 64);
    gather_patches<<<1024, thr, 0, stream>>>(x1b, Ag1, 28, 4, 4, 16);
    gemm_bf16<<<dim3(2, 4), 256, 0, stream>>>(Ag0, sr0wt, nullptr, xr0, 392, 256, 16384, 0);
    gemm_bf16<<<dim3(2, 4), 256, 0, stream>>>(Ag1, sr1wt, nullptr, xr1, 392, 256,  4096, 0);

    // 4. LN + KV proj -> padded [B,NH,112,32] bf16 K/V
    zero_u16<<<128, thr, 0, stream>>>(Kmat, 229376);
    zero_u16<<<128, thr, 0, stream>>>(Vmat, 229376);
    ln_kv<<<392, 256, 0, stream>>>(xr0, sr0_b, norm0_w, norm0_b, kv_w, Kmat, Vmat, 0);
    ln_kv<<<392, 256, 0, stream>>>(xr1, sr1_b, norm1_w, norm1_b, kv_w, Kmat, Vmat, 49);

    // 5. fused attention (245 query tiles x NH x B, one wave each)
    attention<<<dim3(245, 8, 8), 32, 0, stream>>>(Qb0, Qb1, Kmat, Vmat, Xb);

    // 6. output projection (+bias) -> f32 d_out [B,3920,256]
    gemm_bf16<<<dim3(2, 31360 / 128), 256, 0, stream>>>(Xb, pwb, proj_b, (float*)d_out, 31360, 256, 256, 0);
}
